// BidirectionalNeuralSymbolic_7275674599840
// MI455X (gfx1250) — matrix operand order repaired and checked by reading, not compiled, not run on hardware
//
#include <hip/hip_runtime.h>

// Problem constants (match reference)
#define B_   32768
#define DIN_ 1024
#define K_   4096
#define E_   128
#define L_   5
#define H_   512
#define EPS_ 1e-8f

typedef __attribute__((ext_vector_type(16))) __bf16 v16bf;
typedef __attribute__((ext_vector_type(8)))  float  v8f;
typedef __attribute__((ext_vector_type(4)))  float  v4f;

// ---------------------------------------------------------------------------
// WMMA helper: D = A(16x32 bf16) * B(32x16 bf16) + C(16x16 f32)
// ---------------------------------------------------------------------------
__device__ __forceinline__ v8f wmma_bf16(v16bf a, v16bf b, v8f c) {
  return __builtin_amdgcn_wmma_f32_16x16x32_bf16(
      /*neg_a=*/false, a, /*neg_b=*/false, b,
      /*c_mod=*/(short)0, c, /*reuse_a=*/false, /*reuse_b=*/false);
}

// A fragment (16x32, f32 source row-major [M,Kd], converted to bf16).
// Layout: lanes 0-15: M=lane, K = k0..k0+15
//         lanes 16-31: M=lane-16, K = k0+16..k0+31
__device__ __forceinline__ v16bf load_a_f32(const float* __restrict__ A, int Kd,
                                            long m0, int k0, int lane) {
  long row = m0 + (lane & 15);
  int kb = k0 + ((lane & 16) ? 16 : 0);
  const float* p = A + row * (long)Kd + kb;
  v4f f0 = *(const v4f*)(p + 0);
  v4f f1 = *(const v4f*)(p + 4);
  v4f f2 = *(const v4f*)(p + 8);
  v4f f3 = *(const v4f*)(p + 12);
  v16bf a;
#pragma unroll
  for (int i = 0; i < 4; ++i) {
    a[i]      = (__bf16)f0[i];
    a[4 + i]  = (__bf16)f1[i];
    a[8 + i]  = (__bf16)f2[i];
    a[12 + i] = (__bf16)f3[i];
  }
  return a;
}

// B fragment (32x16). Bt is pre-packed bf16 [N, Kd] row-major, so each lane's
// 16 K-values are contiguous. lanes 0-15: N=n0+lane, K=k0..k0+15;
// lanes 16-31: N=n0+lane-16, K=k0+16..k0+31.
__device__ __forceinline__ v16bf load_b_bf16(const __bf16* __restrict__ Bt, int Kd,
                                             int n0, int k0, int lane) {
  const __bf16* p = Bt + (long)(n0 + (lane & 15)) * Kd + k0 + ((lane & 16) ? 16 : 0);
  return *(const v16bf*)p;  // 32B aligned for all our shapes
}

// ---------------------------------------------------------------------------
// Generic GEMM: C[M,N] = act(A[M,Kd] * Bt[N,Kd]^T + bias)
// Per-wave tile 32x64 (2x4 WMMA tiles). 8 waves / block.
// Requires M%32==0, N%64==0, Kd%32==0 and (M/32)*(N/64)%8==0.
// ---------------------------------------------------------------------------
__global__ __launch_bounds__(256) void gemm_bf16_kernel(
    const float* __restrict__ A, const unsigned short* __restrict__ BtRaw,
    float* __restrict__ C, const float* __restrict__ bias,
    int M, int N, int Kd, int act) {
  const __bf16* Bt = (const __bf16*)BtRaw;
  const int lane = threadIdx.x & 31;
  const int wave = threadIdx.x >> 5;
  long tile = (long)blockIdx.x * 8 + wave;
  int ntn = N >> 6;
  long tm = (tile / ntn) * 32;
  int  tn = (int)(tile % ntn) * 64;
  if (tm >= M) return;  // wave-uniform

  v8f acc[2][4] = {};

  for (int k0 = 0; k0 < Kd; k0 += 32) {
    v16bf a0 = load_a_f32(A, Kd, tm,      k0, lane);
    v16bf a1 = load_a_f32(A, Kd, tm + 16, k0, lane);
    v16bf b0 = load_b_bf16(Bt, Kd, tn,      k0, lane);
    v16bf b1 = load_b_bf16(Bt, Kd, tn + 16, k0, lane);
    v16bf b2 = load_b_bf16(Bt, Kd, tn + 32, k0, lane);
    v16bf b3 = load_b_bf16(Bt, Kd, tn + 48, k0, lane);
    acc[0][0] = wmma_bf16(a0, b0, acc[0][0]);
    acc[0][1] = wmma_bf16(a0, b1, acc[0][1]);
    acc[0][2] = wmma_bf16(a0, b2, acc[0][2]);
    acc[0][3] = wmma_bf16(a0, b3, acc[0][3]);
    acc[1][0] = wmma_bf16(a1, b0, acc[1][0]);
    acc[1][1] = wmma_bf16(a1, b1, acc[1][1]);
    acc[1][2] = wmma_bf16(a1, b2, acc[1][2]);
    acc[1][3] = wmma_bf16(a1, b3, acc[1][3]);
  }

  // Epilogue. C/D layout: vgpr r -> row M = r + (lane>=16 ? 8 : 0), col = lane&15
  const int colLane = lane & 15;
  const int rOff = (lane & 16) ? 8 : 0;
#pragma unroll
  for (int j = 0; j < 4; ++j) {
    int col = tn + j * 16 + colLane;
    float bv = bias ? bias[col] : 0.0f;
#pragma unroll
    for (int i = 0; i < 2; ++i) {
      long rowbase = tm + i * 16 + rOff;
#pragma unroll
      for (int r = 0; r < 8; ++r) {
        float v = acc[i][j][r] + bv;
        if (act) v = fmaxf(v, 0.0f);
        C[(rowbase + r) * (long)N + col] = v;
      }
    }
  }
}

// ---------------------------------------------------------------------------
// Fused sims GEMM + scale + softmax + argmax.
// Logits x = 10 * inv_en[m] * inv_pn[n] * (enc_m . proto_n) are bounded in
// [-10,10] (cosine*10), so exp() cannot overflow: softmax computed without
// max-subtraction (shift-invariant, numerically safe here).
// Block = 256 threads (8 waves) handles 16 rows x all K=4096 columns.
// Wave w covers columns [w*512, (w+1)*512).
// Pass 1: WMMA logits -> per-row running sum(exp) + argmax (registers only).
// Reduce across lanes (shuffle) and waves (LDS).
// Pass 2: recompute WMMA logits, write normalized probs once. Only HBM
// traffic on the B*K plane is the single 512MB output write; protos (1MB
// bf16) stays L2-resident.
// ---------------------------------------------------------------------------
__global__ __launch_bounds__(256) void sims_softmax_fused_kernel(
    const float* __restrict__ ENC, const unsigned short* __restrict__ PBraw,
    const float* __restrict__ invE, const float* __restrict__ invP,
    float* __restrict__ probs, int* __restrict__ best_idx) {
  const __bf16* PB = (const __bf16*)PBraw;
  const int lane = threadIdx.x & 31;
  const int wave = threadIdx.x >> 5;
  const long m0 = (long)blockIdx.x * 16;
  const int rOff = (lane & 16) ? 8 : 0;
  const int colLane = lane & 15;
  const int nbase = wave * 512;

  // Preload the wave's A fragments (16 rows x Kd=128) once; reused 256x.
  v16bf afrag[4];
#pragma unroll
  for (int kk = 0; kk < 4; ++kk) afrag[kk] = load_a_f32(ENC, E_, m0, kk * 32, lane);

  // Per-lane row scale factors (this lane touches rows rOff..rOff+7).
  float escale[8];
#pragma unroll
  for (int r = 0; r < 8; ++r) escale[r] = 10.0f * invE[m0 + rOff + r];

  float rsum[8] = {};
  float rmax[8];
  int   ridx[8];
#pragma unroll
  for (int r = 0; r < 8; ++r) { rmax[r] = -INFINITY; ridx[r] = 0x7fffffff; }

  // ---- Pass 1: sums + argmax, no global stores ----
  for (int c = 0; c < 8; ++c) {
    int n0 = nbase + c * 64;
    v8f acc[4] = {};
#pragma unroll
    for (int kk = 0; kk < 4; ++kk) {
#pragma unroll
      for (int j = 0; j < 4; ++j)
        acc[j] = wmma_bf16(afrag[kk],
                           load_b_bf16(PB, E_, n0 + j * 16, kk * 32, lane), acc[j]);
    }
#pragma unroll
    for (int j = 0; j < 4; ++j) {
      int col = n0 + j * 16 + colLane;
      float ip = invP[col];
#pragma unroll
      for (int r = 0; r < 8; ++r) {
        float x = acc[j][r] * escale[r] * ip;
        rsum[r] += __expf(x);
        if (x > rmax[r]) { rmax[r] = x; ridx[r] = col; }  // ascending col/lane
      }
    }
  }

  // Reduce across the 16 lanes sharing each row group (xor masks 1,2,4,8).
#pragma unroll
  for (int r = 0; r < 8; ++r) {
#pragma unroll
    for (int m = 1; m <= 8; m <<= 1) {
      rsum[r] += __shfl_xor(rsum[r], m, 32);
      float om = __shfl_xor(rmax[r], m, 32);
      int   oi = __shfl_xor(ridx[r], m, 32);
      if (om > rmax[r] || (om == rmax[r] && oi < ridx[r])) { rmax[r] = om; ridx[r] = oi; }
    }
  }

  // Cross-wave reduction through LDS.
  __shared__ float ssum[8][16];
  __shared__ float smax[8][16];
  __shared__ int   sidx[8][16];
  __shared__ float sinv[16];
  if (colLane == 0) {  // lane 0 and lane 16 of each wave
#pragma unroll
    for (int r = 0; r < 8; ++r) {
      ssum[wave][rOff + r] = rsum[r];
      smax[wave][rOff + r] = rmax[r];
      sidx[wave][rOff + r] = ridx[r];
    }
  }
  __syncthreads();
  if (threadIdx.x < 16) {
    int row = threadIdx.x;
    float tot = 0.0f, bm = -INFINITY;
    int bi = 0x7fffffff;
#pragma unroll
    for (int w2 = 0; w2 < 8; ++w2) {
      tot += ssum[w2][row];
      float v = smax[w2][row];
      int   ii = sidx[w2][row];
      if (v > bm || (v == bm && ii < bi)) { bm = v; bi = ii; }
    }
    sinv[row] = 1.0f / tot;
    best_idx[m0 + row] = bi;
  }
  __syncthreads();
  float rinv[8];
#pragma unroll
  for (int r = 0; r < 8; ++r) rinv[r] = sinv[rOff + r];

  // ---- Pass 2: recompute logits (compute is ~free vs HBM), write probs ----
  for (int c = 0; c < 8; ++c) {
    int n0 = nbase + c * 64;
    v8f acc[4] = {};
#pragma unroll
    for (int kk = 0; kk < 4; ++kk) {
#pragma unroll
      for (int j = 0; j < 4; ++j)
        acc[j] = wmma_bf16(afrag[kk],
                           load_b_bf16(PB, E_, n0 + j * 16, kk * 32, lane), acc[j]);
    }
#pragma unroll
    for (int j = 0; j < 4; ++j) {
      int col = n0 + j * 16 + colLane;
      float ip = invP[col];
#pragma unroll
      for (int r = 0; r < 8; ++r) {
        float e = __expf(acc[j][r] * escale[r] * ip) * rinv[r];
        probs[(m0 + rOff + r) * (long)K_ + col] = e;
      }
    }
  }
}

// ---------------------------------------------------------------------------
// Weight pre-pack: f32 [R,Ccols] -> bf16 [Ccols,R] (transpose + convert)
// ---------------------------------------------------------------------------
__global__ void cvt_transpose_kernel(const float* __restrict__ in,
                                     unsigned short* __restrict__ outRaw,
                                     int R, int Ccols) {
  __bf16* out = (__bf16*)outRaw;
  long i = (long)blockIdx.x * 256 + threadIdx.x;
  if (i >= (long)R * Ccols) return;
  int r = (int)(i / Ccols), c = (int)(i % Ccols);
  out[(long)c * R + r] = (__bf16)in[i];
}

__global__ void cvt_kernel(const float* __restrict__ in,
                           unsigned short* __restrict__ outRaw, long n) {
  __bf16* out = (__bf16*)outRaw;
  long i = (long)blockIdx.x * 256 + threadIdx.x;
  if (i < n) out[i] = (__bf16)in[i];
}

// ---------------------------------------------------------------------------
// Row inverse-norm for [rows, 128] matrix: inv[r] = 1/max(||row||, EPS)
// One wave per row; shuffle reduction.
// ---------------------------------------------------------------------------
__global__ __launch_bounds__(256) void inv_norm_kernel(
    const float* __restrict__ X, float* __restrict__ inv, int rows) {
  int wave = threadIdx.x >> 5, lane = threadIdx.x & 31;
  long row = (long)blockIdx.x * 8 + wave;
  if (row >= rows) return;
  const float* p = X + row * (long)E_;
  float s = 0.0f;
#pragma unroll
  for (int i = 0; i < E_ / 32; ++i) {
    float v = p[lane + i * 32];
    s += v * v;
  }
#pragma unroll
  for (int m = 16; m; m >>= 1) s += __shfl_xor(s, m, 32);
  if (lane == 0) inv[row] = 1.0f / fmaxf(sqrtf(s), EPS_);
}

// AI[m, 0:128] = encoded[m]; AI[m, 128:256] = protos[best_idx[m]]
__global__ void build_ai_kernel(const float* __restrict__ enc,
                                const float* __restrict__ protos,
                                const int* __restrict__ best,
                                float* __restrict__ AI) {
  long i = (long)blockIdx.x * 256 + threadIdx.x;  // B*256 elements
  long m = i >> 8;
  int  c = (int)(i & 255);
  AI[i] = (c < E_) ? enc[m * E_ + c]
                   : protos[(long)best[m] * E_ + (c - E_)];
}

// hierarchy: out[m,l] = sigmoid(AI[m]·Wh[l] + bh[l]), L=5. Wh cached in LDS.
__global__ __launch_bounds__(256) void hier_kernel(
    const float* __restrict__ AI, const float* __restrict__ Wh,
    const float* __restrict__ bh, float* __restrict__ out) {
  __shared__ float w[L_ * 2 * E_];
  for (int i = threadIdx.x; i < L_ * 2 * E_; i += 256) w[i] = Wh[i];
  __syncthreads();
  long m = (long)blockIdx.x * 256 + threadIdx.x;
  if (m >= B_) return;
  const float* a = AI + m * (long)(2 * E_);
  float acc[L_] = {};
  for (int c = 0; c < 2 * E_; ++c) {
    float av = a[c];
#pragma unroll
    for (int l = 0; l < L_; ++l) acc[l] += av * w[l * 2 * E_ + c];
  }
#pragma unroll
  for (int l = 0; l < L_; ++l)
    out[m * L_ + l] = 1.0f / (1.0f + __expf(-(acc[l] + bh[l])));
}

// ---------------------------------------------------------------------------
extern "C" void kernel_launch(void* const* d_in, const int* in_sizes, int n_in,
                              void* d_out, int out_size, void* d_ws, size_t ws_size,
                              hipStream_t stream) {
  const float* inputs = (const float*)d_in[0];
  const float* W1 = (const float*)d_in[1];
  const float* b1 = (const float*)d_in[2];
  const float* W2 = (const float*)d_in[3];
  const float* b2 = (const float*)d_in[4];
  const float* protos = (const float*)d_in[5];
  const float* W3 = (const float*)d_in[6];
  const float* b3 = (const float*)d_in[7];
  const float* W4 = (const float*)d_in[8];
  const float* b4 = (const float*)d_in[9];
  const float* Wh = (const float*)d_in[10];
  const float* bh = (const float*)d_in[11];

  float* out_probs = (float*)d_out;                              // [B,K]
  float* out_abs   = out_probs + (size_t)B_ * K_;                // [B,E]
  float* out_hier  = out_abs + (size_t)B_ * E_;                  // [B,L]

  // Workspace carve-up
  char* w = (char*)d_ws;
  float* H1  = (float*)w;  w += (size_t)B_ * H_ * 4;             // 64 MB
  float* ENC = (float*)w;  w += (size_t)B_ * E_ * 4;             // 16 MB
  float* AI  = (float*)w;  w += (size_t)B_ * 2 * E_ * 4;         // 32 MB
  float* HB  = (float*)w;  w += (size_t)B_ * E_ * 4;             // 16 MB
  unsigned short* W1t = (unsigned short*)w; w += (size_t)H_ * DIN_ * 2;
  unsigned short* W2t = (unsigned short*)w; w += (size_t)E_ * H_ * 2;
  unsigned short* W3t = (unsigned short*)w; w += (size_t)E_ * 2 * E_ * 2;
  unsigned short* W4t = (unsigned short*)w; w += (size_t)E_ * E_ * 2;
  unsigned short* PB  = (unsigned short*)w; w += (size_t)K_ * E_ * 2;
  float* invE = (float*)w; w += (size_t)B_ * 4;
  float* invP = (float*)w; w += (size_t)K_ * 4;
  int*   BI   = (int*)w;

  // 1) Pre-pack weights to bf16 [N,K] layout
  cvt_transpose_kernel<<<(DIN_ * H_ + 255) / 256, 256, 0, stream>>>(W1, W1t, DIN_, H_);
  cvt_transpose_kernel<<<(H_ * E_ + 255) / 256, 256, 0, stream>>>(W2, W2t, H_, E_);
  cvt_transpose_kernel<<<(2 * E_ * E_ + 255) / 256, 256, 0, stream>>>(W3, W3t, 2 * E_, E_);
  cvt_transpose_kernel<<<(E_ * E_ + 255) / 256, 256, 0, stream>>>(W4, W4t, E_, E_);
  cvt_kernel<<<(K_ * E_ + 255) / 256, 256, 0, stream>>>(protos, PB, (long)K_ * E_);

  // 2) Encoder: H1 = relu(X @ W1 + b1); ENC = H1 @ W2 + b2
  {
    int blocks = (B_ / 32) * (H_ / 64) / 8;  // 1024
    gemm_bf16_kernel<<<blocks, 256, 0, stream>>>(inputs, W1t, H1, b1, B_, H_, DIN_, 1);
  }
  {
    int blocks = (B_ / 32) * (E_ / 64) / 8;  // 256
    gemm_bf16_kernel<<<blocks, 256, 0, stream>>>(H1, W2t, ENC, b2, B_, E_, H_, 0);
  }

  // 3) Row norms (cosine-sim scaling factors)
  inv_norm_kernel<<<(B_ + 7) / 8, 256, 0, stream>>>(ENC, invE, B_);
  inv_norm_kernel<<<(K_ + 7) / 8, 256, 0, stream>>>(protos, invP, K_);

  // 4+5) Fused sims GEMM + softmax + argmax: single 512MB write, no re-read.
  sims_softmax_fused_kernel<<<B_ / 16, 256, 0, stream>>>(ENC, PB, invE, invP,
                                                         out_probs, BI);

  // 6) AI = [encoded, protos[best_idx]]
  build_ai_kernel<<<B_, 256, 0, stream>>>(ENC, protos, BI, AI);

  // 7) Abstraction: HB = relu(AI @ W3 + b3); out_abs = HB @ W4 + b4
  {
    int blocks = (B_ / 32) * (E_ / 64) / 8;  // 256
    gemm_bf16_kernel<<<blocks, 256, 0, stream>>>(AI, W3t, HB, b3, B_, E_, 2 * E_, 1);
    gemm_bf16_kernel<<<blocks, 256, 0, stream>>>(HB, W4t, out_abs, b4, B_, E_, E_, 0);
  }

  // 8) Hierarchy scores
  hier_kernel<<<B_ / 256, 256, 0, stream>>>(AI, Wh, bh, out_hier);
}
